// HierarchicalReasoningModel_46145128628395
// MI455X (gfx1250) — matrix operand order repaired and checked by reading, not compiled, not run on hardware
//
#include <hip/hip_runtime.h>

// ---------------------------------------------------------------------------
// Types for CDNA5 WMMA (wave32): v_wmma_f32_16x16x32_bf16
// ---------------------------------------------------------------------------
typedef __attribute__((ext_vector_type(16))) __bf16 bf16x16;
typedef __attribute__((ext_vector_type(8)))  __bf16 bf16x8;
typedef __attribute__((ext_vector_type(8)))  float  f32x8;
typedef unsigned short u16;

__device__ __forceinline__ bf16x16 frag_cat(bf16x8 lo, bf16x8 hi) {
  return __builtin_shufflevector(lo, hi, 0,1,2,3,4,5,6,7,8,9,10,11,12,13,14,15);
}
// f32 <-> bf16 via bit ops (round-to-nearest-even); avoids scalar __bf16 math.
__device__ __forceinline__ u16 f2b(float f) {
  unsigned u = __float_as_uint(f);
  unsigned r = (u + 0x7FFFu + ((u >> 16) & 1u)) >> 16;
  return (u16)r;
}
__device__ __forceinline__ float b2f(u16 h) {
  return __uint_as_float(((unsigned)h) << 16);
}
__device__ __forceinline__ float sigmoidf(float x) { return 1.0f / (1.0f + __expf(-x)); }

// ---------------------------------------------------------------------------
// Conversions / init
// ---------------------------------------------------------------------------
__global__ void k_f32_to_bf16(const float* __restrict__ s, u16* __restrict__ d, int n) {
  int i = blockIdx.x * 256 + threadIdx.x;
  if (i < n) d[i] = f2b(s[i]);
}
__global__ void k_zero_bf16(u16* __restrict__ d, int n) {
  int i = blockIdx.x * 256 + threadIdx.x;
  if (i < n) d[i] = 0;
}

// ---------------------------------------------------------------------------
// Generic bf16 WMMA GEMM:  O(M,N) = A(M,K) @ W(N,K)^T  [+bias0+bias1][+addF][+addB]
// A bf16 row-major, W bf16 row-major (PyTorch (out,in) layout), f32 accumulate.
// Register blocking: each wave computes 1 M-tile x 4 N-tiles (16x64 output),
// sharing one A fragment across 4 WMMAs per k-step (2.5 b128 loads / WMMA).
// 8 waves (256 threads) per block. Requires N % 64 == 0, K % 32 == 0, M % 16 == 0.
// ---------------------------------------------------------------------------
__global__ void k_gemm(const u16* __restrict__ A, const u16* __restrict__ W,
                       const float* __restrict__ bias0, const float* __restrict__ bias1,
                       const float* __restrict__ addF, const u16* __restrict__ addB,
                       float* __restrict__ outF, u16* __restrict__ outB,
                       int M, int N, int K, int relu) {
  const int lane   = threadIdx.x & 31;
  const int wave   = threadIdx.x >> 5;
  const int tile   = blockIdx.x * 8 + wave;
  const int nblks  = N >> 6;                 // N-blocks of 64
  const int mt     = tile / nblks;
  const int ntb    = tile - mt * nblks;
  if (mt >= (M >> 4)) return;                // wave-uniform exit (EXEC stays all-ones)
  const int khalf = lane >> 4;
  const int mr    = lane & 15;
  const u16* Ar = A + (size_t)(mt * 16 + mr) * K;
  const u16* Wr0 = W + (size_t)(ntb * 64 + mr) * K;          // n-tile q row: +q*16*K
  f32x8 acc0 = {}, acc1 = {}, acc2 = {}, acc3 = {};
  for (int k0 = 0; k0 < K; k0 += 32) {
    // A frag: lane (m=mr) needs K = k0 + khalf*16 + {0..15} (contiguous, matches B)
    bf16x8 alo = *(const bf16x8*)(Ar + k0 + khalf * 16);
    bf16x8 ahi = *(const bf16x8*)(Ar + k0 + khalf * 16 + 8);
    bf16x16 af = frag_cat(alo, ahi);
    // B frags: lane (n=mr) needs K = k0 + khalf*16 + {0..15}, contiguous in W's (N,K) rows
    const size_t wko = (size_t)k0 + khalf * 16;
    bf16x16 b0 = *(const bf16x16*)(Wr0 + wko);
    bf16x16 b1 = *(const bf16x16*)(Wr0 + (size_t)16 * K + wko);
    bf16x16 b2 = *(const bf16x16*)(Wr0 + (size_t)32 * K + wko);
    bf16x16 b3 = *(const bf16x16*)(Wr0 + (size_t)48 * K + wko);
    acc0 = __builtin_amdgcn_wmma_f32_16x16x32_bf16(false, af, false, b0, (short)0, acc0, false, false);
    acc1 = __builtin_amdgcn_wmma_f32_16x16x32_bf16(false, af, false, b1, (short)0, acc1, false, false);
    acc2 = __builtin_amdgcn_wmma_f32_16x16x32_bf16(false, af, false, b2, (short)0, acc2, false, false);
    acc3 = __builtin_amdgcn_wmma_f32_16x16x32_bf16(false, af, false, b3, (short)0, acc3, false, false);
  }
  f32x8 accs[4] = {acc0, acc1, acc2, acc3};
#pragma unroll
  for (int q = 0; q < 4; ++q) {
    const int n = ntb * 64 + q * 16 + mr;
    float bb = 0.f;
    if (bias0) bb += bias0[n];
    if (bias1) bb += bias1[n];
#pragma unroll
    for (int v = 0; v < 8; ++v) {
      int m = mt * 16 + v + khalf * 8;       // C/D layout: VGPR v -> row v + 8*(lane>>4)
      size_t idx = (size_t)m * N + n;
      float val = accs[q][v] + bb;
      if (addF) val += addF[idx];
      if (addB) val += b2f(addB[idx]);
      if (relu) val = fmaxf(val, 0.f);
      if (outF) outF[idx] = val;
      if (outB) outB[idx] = f2b(val);
    }
  }
}

// ---------------------------------------------------------------------------
// Device-scope grid barrier across the 8 persistent workgroups.
// ---------------------------------------------------------------------------
__device__ __forceinline__ void grid_sync(unsigned* bar, unsigned phase) {
  __syncthreads();
  __threadfence();
  if (threadIdx.x == 0) {
    __hip_atomic_fetch_add(bar, 1u, __ATOMIC_RELEASE, __HIP_MEMORY_SCOPE_AGENT);
    const unsigned target = 8u * phase;
    while (__hip_atomic_load(bar, __ATOMIC_ACQUIRE, __HIP_MEMORY_SCOPE_AGENT) < target)
      __builtin_amdgcn_s_sleep(1);
  }
  __syncthreads();
}

// ---------------------------------------------------------------------------
// Persistent LSTM recurrence for one layer (all 64 timesteps in one launch).
// Grid = 8 WGs x 256 threads. WG w owns hidden slice [w*64, w*64+64): that is
// 4 gate blocks x 4 N-tiles = 16 WMMA tiles, 2 per wave. Gates staged in LDS,
// cell state lives in LDS, h double-buffered in global, one grid sync per step.
// Whh (2048,512) bf16 streams from L2 each step (~2MB, L2-resident).
// ---------------------------------------------------------------------------
__global__ void k_lstm_seq(const u16* __restrict__ Whh,   // (2048,512) bf16 row-major
                           const float* __restrict__ xg,  // (1024,2048) f32, row = b*64+t
                           u16* __restrict__ y,           // (1024,512) bf16, row = b*64+t
                           u16* __restrict__ hbuf,        // 2 x (16,512) bf16
                           unsigned* __restrict__ bar) {
  __shared__ float gl[4][64][16];   // [gate][j_local][batch] f32 gates (16 KB)
  __shared__ float cl[64][16];      // cell state f32 (4 KB)
  const int w     = blockIdx.x;     // 0..7 hidden slice
  const int lane  = threadIdx.x & 31;
  const int wave  = threadIdx.x >> 5;
  const int khalf = lane >> 4;
  const int mr    = lane & 15;

  for (int i = threadIdx.x; i < 64 * 16; i += 256) ((float*)cl)[i] = 0.f;
  for (int i = threadIdx.x; i < 16 * 64; i += 256) {
    int bb = i >> 6, j = i & 63;
    hbuf[(size_t)bb * 512 + w * 64 + j] = 0;   // h0 = 0 in buffer 0
  }
  unsigned phase = 1;
  grid_sync(bar, phase++);                      // everyone sees zeroed h0

  const int tA = wave * 2, tB = wave * 2 + 1;   // this wave's two tiles
  const int gA = tA >> 2, fA = tA & 3;
  const int gB = tB >> 2, fB = tB & 3;
  const int n0A = gA * 512 + w * 64 + fA * 16;
  const int n0B = gB * 512 + w * 64 + fB * 16;
  const u16* WA = Whh + (size_t)(n0A + mr) * 512;
  const u16* WB = Whh + (size_t)(n0B + mr) * 512;

  for (int t = 0; t < 64; ++t) {
    const u16* hprev = hbuf + ((t & 1) ? 16 * 512 : 0);
    u16* hnext       = hbuf + ((t & 1) ? 0 : 16 * 512);
    f32x8 accA, accB;
#pragma unroll
    for (int v = 0; v < 8; ++v) {               // init acc from precomputed x-gates
      int bb = v + khalf * 8;
      size_t row = (size_t)(bb * 64 + t) * 2048;
      accA[v] = xg[row + n0A + mr];
      accB[v] = xg[row + n0B + mr];
    }
    const u16* Ar = hprev + (size_t)mr * 512;
#pragma unroll 4
    for (int k0 = 0; k0 < 512; k0 += 32) {
      bf16x8 alo = *(const bf16x8*)(Ar + k0 + khalf * 16);
      bf16x8 ahi = *(const bf16x8*)(Ar + k0 + khalf * 16 + 8);
      bf16x16 af = frag_cat(alo, ahi);
      bf16x16 b0 = *(const bf16x16*)(WA + k0 + khalf * 16);
      accA = __builtin_amdgcn_wmma_f32_16x16x32_bf16(false, af, false, b0,
                                                     (short)0, accA, false, false);
      bf16x16 b1 = *(const bf16x16*)(WB + k0 + khalf * 16);
      accB = __builtin_amdgcn_wmma_f32_16x16x32_bf16(false, af, false, b1,
                                                     (short)0, accB, false, false);
    }
#pragma unroll
    for (int v = 0; v < 8; ++v) {               // stage gates in LDS
      int bb = v + khalf * 8;
      gl[gA][fA * 16 + mr][bb] = accA[v];
      gl[gB][fB * 16 + mr][bb] = accB[v];
    }
    __syncthreads();
    for (int i = threadIdx.x; i < 64 * 16; i += 256) {   // cell update (i,f,g,o)
      int j = i >> 4, bb = i & 15;
      float ig = sigmoidf(gl[0][j][bb]);
      float fg = sigmoidf(gl[1][j][bb]);
      float gg = tanhf(gl[2][j][bb]);
      float og = sigmoidf(gl[3][j][bb]);
      float c  = fg * cl[j][bb] + ig * gg;
      cl[j][bb] = c;
      u16 h = f2b(og * tanhf(c));
      hnext[(size_t)bb * 512 + w * 64 + j] = h;
      y[(size_t)(bb * 64 + t) * 512 + w * 64 + j] = h;
    }
    grid_sync(bar, phase++);                    // h(t) visible before t+1
  }
}

// ---------------------------------------------------------------------------
// LayerNorm over concat([low, high]) (1024 features), out bf16 for W1 GEMM.
// ---------------------------------------------------------------------------
__global__ void k_ln(const u16* __restrict__ low, const u16* __restrict__ high,
                     const float* __restrict__ g, const float* __restrict__ b,
                     u16* __restrict__ out) {
  __shared__ float s1[256], s2[256];
  int r = blockIdx.x;
  float a = 0.f, q = 0.f;
  for (int c = threadIdx.x; c < 1024; c += 256) {
    float v = (c < 512) ? b2f(low[(size_t)r * 512 + c]) : b2f(high[(size_t)r * 512 + c - 512]);
    a += v; q += v * v;
  }
  s1[threadIdx.x] = a; s2[threadIdx.x] = q;
  __syncthreads();
  for (int o = 128; o > 0; o >>= 1) {
    if (threadIdx.x < o) { s1[threadIdx.x] += s1[threadIdx.x + o]; s2[threadIdx.x] += s2[threadIdx.x + o]; }
    __syncthreads();
  }
  float mu  = s1[0] * (1.f / 1024.f);
  float var = s2[0] * (1.f / 1024.f) - mu * mu;
  float inv = rsqrtf(var + 1e-5f);
  for (int c = threadIdx.x; c < 1024; c += 256) {
    float v = (c < 512) ? b2f(low[(size_t)r * 512 + c]) : b2f(high[(size_t)r * 512 + c - 512]);
    out[(size_t)r * 1024 + c] = f2b((v - mu) * inv * g[c] + b[c]);
  }
}

// Final tiny GEMM: out(1024,10) = h2(1024,256) @ W3(10,256)^T + b3 (f32)
__global__ void k_head3(const u16* __restrict__ h2, const float* __restrict__ W3,
                        const float* __restrict__ b3, float* __restrict__ out) {
  int idx = blockIdx.x * 256 + threadIdx.x;
  if (idx >= 1024 * 10) return;
  int r = idx / 10, o = idx - r * 10;
  float acc = b3[o];
  for (int k = 0; k < 256; ++k) acc += b2f(h2[(size_t)r * 256 + k]) * W3[o * 256 + k];
  out[idx] = acc;
}

// ---------------------------------------------------------------------------
// Host orchestration
// ---------------------------------------------------------------------------
extern "C" void kernel_launch(void* const* d_in, const int* in_sizes, int n_in,
                              void* d_out, int out_size, void* d_ws, size_t ws_size,
                              hipStream_t stream) {
  (void)in_sizes; (void)n_in; (void)out_size; (void)ws_size;
  const float* x        = (const float*)d_in[0];
  const float* Wp_in    = (const float*)d_in[1];
  const float* bp_in    = (const float*)d_in[2];
  const float* low_Wih  = (const float*)d_in[3];
  const float* low_Whh  = (const float*)d_in[4];
  const float* low_bih  = (const float*)d_in[5];
  const float* low_bhh  = (const float*)d_in[6];
  const float* high_Wih = (const float*)d_in[7];
  const float* high_Whh = (const float*)d_in[8];
  const float* high_bih = (const float*)d_in[9];
  const float* high_bhh = (const float*)d_in[10];
  const float* W_lowproj  = (const float*)d_in[11];
  const float* W_highproj = (const float*)d_in[12];
  const float* ln_g = (const float*)d_in[13];
  const float* ln_b = (const float*)d_in[14];
  const float* W1 = (const float*)d_in[15];
  const float* b1 = (const float*)d_in[16];
  const float* W2 = (const float*)d_in[17];
  const float* b2 = (const float*)d_in[18];
  const float* W3 = (const float*)d_in[19];
  const float* b3 = (const float*)d_in[20];

  const int BS = 1024, H = 512, G = 2048;     // B*S, hidden, 4*H
  char* base = (char*)d_ws;
  size_t off = 0;
  auto alloc = [&](size_t bytes) -> char* {
    char* p = base + off;
    off += (bytes + 255) & ~(size_t)255;
    return p;
  };
  u16* bx    = (u16*)alloc((size_t)BS * H * 2);
  u16* bWp   = (u16*)alloc((size_t)H * H * 2);
  u16* bLWih = (u16*)alloc((size_t)4 * G * H * 2);
  u16* bLWhh = (u16*)alloc((size_t)4 * G * H * 2);
  u16* bHWih = (u16*)alloc((size_t)4 * G * H * 2);
  u16* bHWhh = (u16*)alloc((size_t)4 * G * H * 2);
  u16* bWlp  = (u16*)alloc((size_t)H * H * 2);
  u16* bWhp  = (u16*)alloc((size_t)H * H * 2);
  u16* bW1   = (u16*)alloc((size_t)512 * 1024 * 2);
  u16* bW2   = (u16*)alloc((size_t)256 * 512 * 2);
  float* emb = (float*)alloc((size_t)BS * H * 4);
  float* xg  = (float*)alloc((size_t)BS * G * 4);
  u16* combined = (u16*)alloc((size_t)BS * H * 2);
  u16* actA  = (u16*)alloc((size_t)BS * H * 2);
  u16* actB  = (u16*)alloc((size_t)BS * H * 2);
  u16* low_buf  = (u16*)alloc((size_t)BS * H * 2);
  u16* high_buf = (u16*)alloc((size_t)BS * H * 2);
  u16* hbuf  = (u16*)alloc((size_t)2 * 16 * 512 * 2);
  u16* normed = (u16*)alloc((size_t)BS * 1024 * 2);
  u16* h1    = (u16*)alloc((size_t)BS * 512 * 2);
  u16* h2    = (u16*)alloc((size_t)BS * 256 * 2);
  unsigned* bars = (unsigned*)alloc(64 * 64);   // one 64B slot per recurrence launch

  hipMemsetAsync(bars, 0, 64 * 64, stream);

  auto conv = [&](const float* s, u16* d, int n) {
    k_f32_to_bf16<<<(n + 255) / 256, 256, 0, stream>>>(s, d, n);
  };
  auto zero = [&](u16* d, int n) {
    k_zero_bf16<<<(n + 255) / 256, 256, 0, stream>>>(d, n);
  };
  auto gemm = [&](const u16* A, const u16* W, const float* bias0, const float* bias1,
                  const float* addF, const u16* addB, float* outF, u16* outB,
                  int M, int N, int K, int relu) {
    int tiles = (M >> 4) * (N >> 6);           // 16x64 output blocks per wave
    k_gemm<<<(tiles + 7) / 8, 256, 0, stream>>>(A, W, bias0, bias1, addF, addB,
                                                outF, outB, M, N, K, relu);
  };

  conv(x, bx, BS * H);
  conv(Wp_in, bWp, H * H);
  conv(low_Wih, bLWih, 4 * G * H);
  conv(low_Whh, bLWhh, 4 * G * H);
  conv(high_Wih, bHWih, 4 * G * H);
  conv(high_Whh, bHWhh, 4 * G * H);
  conv(W_lowproj, bWlp, H * H);
  conv(W_highproj, bWhp, H * H);
  conv(W1, bW1, 512 * 1024);
  conv(W2, bW2, 256 * 512);
  zero(low_buf, BS * H);
  zero(high_buf, BS * H);

  // emb = x @ Wp_in^T + bp_in  (kept f32; only ever consumed as an addend)
  gemm(bx, bWp, bp_in, nullptr, nullptr, nullptr, emb, nullptr, BS, H, H, 0);

  int barIdx = 0;
  // level_step: combined = first + second @ Wproj^T + emb; then 4 stacked LSTM layers
  auto level_step = [&](u16* first, u16* second, const u16* Wproj,
                        const u16* Wih_b, const u16* Whh_b,
                        const float* bih, const float* bhh, u16* dest) {
    gemm(second, Wproj, nullptr, nullptr, emb, first, nullptr, combined, BS, H, H, 0);
    u16* in = combined;
    for (int l = 0; l < 4; ++l) {
      gemm(in, Wih_b + (size_t)l * G * H, bih + l * G, bhh + l * G,
           nullptr, nullptr, xg, nullptr, BS, G, H, 0);
      u16* out = (l == 3) ? dest : ((l & 1) ? actB : actA);
      k_lstm_seq<<<8, 256, 0, stream>>>(Whh_b + (size_t)l * G * H, xg, out, hbuf,
                                        bars + 16 * (barIdx++));
      in = out;
    }
  };
  auto low_step = [&]() {
    level_step(low_buf, high_buf, bWlp, bLWih, bLWhh, low_bih, low_bhh, low_buf);
  };
  auto high_step = [&]() {
    level_step(high_buf, low_buf, bWhp, bHWih, bHWhh, high_bih, high_bhh, high_buf);
  };

  for (int step = 0; step < 7; ++step) {   // NSTEP*TSTEP - 1
    low_step();
    if ((step + 1) % 4 == 0) high_step();
  }
  low_step();    // final low
  high_step();   // final high

  // Head: LN(concat) -> W1+relu -> W2+relu -> W3
  k_ln<<<1024, 256, 0, stream>>>(low_buf, high_buf, ln_g, ln_b, normed);
  gemm(normed, bW1, b1, nullptr, nullptr, nullptr, nullptr, h1, BS, 512, 1024, 1);
  gemm(h1, bW2, b2, nullptr, nullptr, nullptr, nullptr, h2, BS, 256, 512, 1);
  k_head3<<<(1024 * 10 + 255) / 256, 256, 0, stream>>>(h2, W3, b3, (float*)d_out);
}
